// CannyDetector_6940667150754
// MI455X (gfx1250) — matrix operand order repaired and checked by reading, not compiled, not run on hardware
//
#include <hip/hip_runtime.h>
#include <hip/hip_bf16.h>
#include <stdint.h>

typedef __attribute__((ext_vector_type(16))) _Float16 v16h;
typedef __attribute__((ext_vector_type(8)))  float    v8f;

#define IMG_H 1024
#define IMG_W 1024

// Fused Canny: gaussian(WMMA) -> sobel -> mag/ori -> NMS -> hysteresis(1 step)
__global__ void __launch_bounds__(256)
canny_fused_kernel(const float* __restrict__ img, float* __restrict__ out)
{
    // LDS staging buffers (total ~51.8 KB of the 320 KB/WGP)
    __shared__ _Float16      sIn[3][42][44];   // input tile + halo5, f16
    __shared__ _Float16      sHh[3][48][50];   // horizontal-blur result (WMMA-padded), f16
    __shared__ float         sBl[3][38][40];   // separable blur result, f32
    __shared__ float         sMag[36][38];     // per-pixel gradient magnitude (3-ch sum)
    __shared__ unsigned char sDir[36][38];     // quantized orientation index (0..7)
    __shared__ unsigned char sSt[34][36];      // bit0=strong(>100), bit1=weak(10..100)

    const int tid  = threadIdx.x;
    const int lane = tid & 31;     // wave32 lane
    const int wave = tid >> 5;     // 8 waves per block
    const int X0 = blockIdx.x * 32;
    const int Y0 = blockIdx.y * 32;
    const int b  = blockIdx.z;

    const float gf[5] = {0.13533528323661270f, 0.60653065971263342f, 1.0f,
                         0.60653065971263342f, 0.13533528323661270f};

    // ---- S0: load 42x42 x 3ch input region (zero-padded), convert to f16 ----
    for (int i = tid; i < 3 * 42 * 42; i += 256) {
        int ch = i / (42 * 42), rem = i % (42 * 42);
        int r = rem / 42, c = rem % 42;
        int gy = Y0 - 5 + r, gx = X0 - 5 + c;
        float v = 0.0f;
        if ((unsigned)gy < (unsigned)IMG_H && (unsigned)gx < (unsigned)IMG_W)
            v = img[(((size_t)b * 3 + ch) * IMG_H + gy) * IMG_W + gx];
        sIn[ch][r][c] = (_Float16)v;
    }
    __syncthreads();

    // ---- S1: horizontal gaussian via WMMA:  H[h][hc] = sum_j g[j] * in[h][hc+j]
    // Per 16x16 tile: D = A(data 16x32) x B(band coeffs 32x16), 27 tile-jobs (3ch x 3x3)
    for (int job = wave; job < 27; job += 8) {
        int ch = job / 9, t = job % 9, ti = t / 3, tj = t % 3;
        int MN  = lane & 15;
        int kbA = (lane < 16) ? 0 : 16;  // A: contiguous K halves across lane groups
        int kbB = (lane < 16) ? 0 : 16;  // B: contiguous K halves across lane groups
        int arow = 16 * ti + MN; if (arow > 41) arow = 41;
        v16h A, Bf;
#pragma unroll
        for (int e = 0; e < 16; ++e) {
            int kA = kbA + e;                      // K: {0-15} / {16-31}
            int acol = 16 * tj + kA; if (acol > 41) acol = 41;   // clamp: finite
            A[e] = sIn[ch][arow][acol];
            int kB = kbB + e;
            int d  = kB - MN;                      // band: Wh[k][n] = g[k-n]
            Bf[e] = (d >= 0 && d < 5) ? (_Float16)gf[d] : (_Float16)0.0f;
        }
        v8f C = {};
        C = __builtin_amdgcn_wmma_f32_16x16x32_f16(false, A, false, Bf,
                                                   (short)0, C, false, false);
        int mo = (lane < 16) ? 0 : 8;              // documented C/D layout
#pragma unroll
        for (int v = 0; v < 8; ++v)
            sHh[ch][16 * ti + mo + v][16 * tj + MN] = (_Float16)C[v];
    }
    __syncthreads();

    // ---- S2: vertical gaussian via WMMA:  Bl[r][c] = sum_i g[i] * H[r+i][c]
    // Per 16x16 tile: D = A(band coeffs 16x32) x B(H data 32x16)
    for (int job = wave; job < 27; job += 8) {
        int ch = job / 9, t = job % 9, ti = t / 3, tj = t % 3;
        int MN  = lane & 15;
        int kbA = (lane < 16) ? 0 : 16;
        int kbB = (lane < 16) ? 0 : 16;
        v16h A, Bf;
#pragma unroll
        for (int e = 0; e < 16; ++e) {
            int kA = kbA + e;
            int d  = kA - MN;                      // band: Wv[m][k] = g[k-m]
            A[e] = (d >= 0 && d < 5) ? (_Float16)gf[d] : (_Float16)0.0f;
            int kB   = kbB + e;
            int hrow = 16 * ti + kB; if (hrow > 47) hrow = 47;   // clamp: finite
            Bf[e] = sHh[ch][hrow][16 * tj + MN];
        }
        v8f C = {};
        C = __builtin_amdgcn_wmma_f32_16x16x32_f16(false, A, false, Bf,
                                                   (short)0, C, false, false);
        int mo = (lane < 16) ? 0 : 8;
#pragma unroll
        for (int v = 0; v < 8; ++v) {
            int r = 16 * ti + mo + v, c = 16 * tj + MN;
            if (r < 38 && c < 38) sBl[ch][r][c] = C[v];
        }
    }
    __syncthreads();

    // ---- S3: sobel + magnitude (3-ch sum) + quantized orientation (36x36) ----
    for (int i = tid; i < 36 * 36; i += 256) {
        int r = i / 36, c = i % 36;
        int br = r + 1, bc = c + 1;                // blur-local coords
        float sgx = 0.0f, sgy = 0.0f, mag = 0.0f;
#pragma unroll
        for (int ch = 0; ch < 3; ++ch) {
            float a00 = sBl[ch][br - 1][bc - 1], a01 = sBl[ch][br - 1][bc], a02 = sBl[ch][br - 1][bc + 1];
            float a10 = sBl[ch][br    ][bc - 1],                            a12 = sBl[ch][br    ][bc + 1];
            float a20 = sBl[ch][br + 1][bc - 1], a21 = sBl[ch][br + 1][bc], a22 = sBl[ch][br + 1][bc + 1];
            float gx = (a00 - a02) + 2.0f * (a10 - a12) + (a20 - a22);
            float gy = (a00 - a20) + 2.0f * (a01 - a21) + (a02 - a22);
            sgx += gx; sgy += gy;
            mag += sqrtf(gx * gx + gy * gy);
        }
        sMag[r][c] = mag;
        // ori/45 = atan2*(4/pi)+4 in [0,8]; round-nearest-even then mod 8
        float o = atan2f(sgy, sgx) * 1.27323954473516f + 4.0f;
        sDir[r][c] = (unsigned char)(__float2int_rn(o) & 7);
    }
    __syncthreads();

    // ---- S4: NMS + double threshold (34x34) ----
    const int dyo[8] = {0, 1, 1, 1, 0, -1, -1, -1};
    const int dxo[8] = {1, 1, 0, -1, -1, -1, 0, 1};
    for (int i = tid; i < 34 * 34; i += 256) {
        int s = i / 34, u = i % 34;
        int r = s + 1, c = u + 1;                  // mag-local coords
        float m = sMag[r][c];
        int k  = sDir[r][c];
        int dy = dyo[k], dx = dxo[k];
        float sp = m - sMag[r + dy][c + dx];       // sel_pos
        float sn = m - sMag[r - dy][c - dx];       // sel_neg (opposite dir)
        float thin = (sp > 0.0f && sn > 0.0f) ? m : 0.0f;
        unsigned char st = 0;
        if (thin > 100.0f)      st = 1;            // strong
        else if (thin >= 10.0f) st = 2;            // weak [10..100]
        sSt[s][u] = st;
    }
    __syncthreads();

    // ---- S5: 8-connected hysteresis step + border zero + store (32x32) ----
    for (int i = tid; i < 32 * 32; i += 256) {
        int oy = i / 32, ox = i % 32;
        int s = oy + 1, u = ox + 1;                // state-local coords
        unsigned char st = sSt[s][u];
        int edge = st & 1;
        if (!edge && (st & 2)) {
            int any = (sSt[s - 1][u - 1] & 1) | (sSt[s - 1][u] & 1) | (sSt[s - 1][u + 1] & 1)
                    | (sSt[s    ][u - 1] & 1) |                       (sSt[s    ][u + 1] & 1)
                    | (sSt[s + 1][u - 1] & 1) | (sSt[s + 1][u] & 1) | (sSt[s + 1][u + 1] & 1);
            edge = any;
        }
        int gy = Y0 + oy, gx = X0 + ox;
        if (gy == 0 || gy == IMG_H - 1 || gx == 0 || gx == IMG_W - 1) edge = 0;
        out[((size_t)b * IMG_H + gy) * IMG_W + gx] = edge ? 1.0f : 0.0f;
    }
}

extern "C" void kernel_launch(void* const* d_in, const int* in_sizes, int n_in,
                              void* d_out, int out_size, void* d_ws, size_t ws_size,
                              hipStream_t stream)
{
    // setup_inputs order: img, gauss_h, gauss_v, sobel_h, sobel_v, dir_filt, conn_filt.
    // Filters are fixed constants of the reference; they are hardcoded in-kernel.
    const float* img = (const float*)d_in[0];
    float* out = (float*)d_out;
    dim3 grid(IMG_W / 32, IMG_H / 32, 8);
    canny_fused_kernel<<<grid, 256, 0, stream>>>(img, out);
    (void)in_sizes; (void)n_in; (void)out_size; (void)d_ws; (void)ws_size;
}
